// Attention_23613730194049
// MI455X (gfx1250) — matrix operand order repaired and checked by reading, not compiled, not run on hardware
//
#include <hip/hip_runtime.h>

typedef __bf16 bf16_t;
typedef __attribute__((ext_vector_type(16))) __bf16 v16bf;
typedef __attribute__((ext_vector_type(8)))  __bf16 v8bf;
typedef __attribute__((ext_vector_type(8)))  float  v8f;
typedef __attribute__((ext_vector_type(4))) unsigned u32x4;
typedef __attribute__((ext_vector_type(4))) int      i32x4;
typedef __attribute__((ext_vector_type(8))) int      i32x8;

#define BATCH 8
#define SEQ   2048
#define DIM   512
#define ROWS  (BATCH * SEQ)   // 16384

// ---------------------------------------------------------------------------
// WMMA helpers (CDNA5 wave32, 16x16x32 bf16 -> f32)
// ---------------------------------------------------------------------------
static __device__ __forceinline__ v8f wmma_bf16(v16bf a, v16bf b, v8f c) {
  // args: (neg_a, A, neg_b, B, c_mod, C, reuse_a, reuse_b)
  return __builtin_amdgcn_wmma_f32_16x16x32_bf16(false, a, false, b, (short)0, c,
                                                 false, false);
}

static __device__ __forceinline__ v16bf wmma_cat(v8bf lo, v8bf hi) {
  v16bf r;
#pragma unroll
  for (int i = 0; i < 8; ++i) { r[i] = lo[i]; r[i + 8] = hi[i]; }
  return r;
}

static __device__ __forceinline__ v8f vzero8() {
  v8f z;
#pragma unroll
  for (int i = 0; i < 8; ++i) z[i] = 0.0f;
  return z;
}

// ---------------------------------------------------------------------------
// Tensor Data Mover: DMA a [rows x cols] f32 tile (row stride = stride_elts)
// from global memory into LDS at byte offset lds_off. 2D descriptor per
// CDNA5 ISA (type=2, data_size=4B, count=1).
// ---------------------------------------------------------------------------
static __device__ __forceinline__ void tdm_load_2d_f32(const void* gaddr,
                                                       unsigned lds_off,
                                                       int rows, int cols,
                                                       int stride_elts) {
  unsigned long long ga = (unsigned long long)gaddr;
  u32x4 g0;
  g0[0] = 1u;                                   // count=1, flags=0
  g0[1] = lds_off;                              // lds_addr (bytes)
  g0[2] = (unsigned)ga;                         // global_addr[31:0]
  g0[3] = (unsigned)((ga >> 32) & 0x01ffffffull) | (2u << 30);  // addr[56:32] | type=2
  i32x8 g1;
  g1[0] = (2 << 16);                            // data_size = 4 bytes
  g1[1] = (stride_elts & 0xffff) << 16;         // tensor_dim0[15:0]
  g1[2] = ((stride_elts >> 16) & 0xffff) | ((SEQ & 0xffff) << 16); // dim0 hi | dim1 lo
  g1[3] = (cols & 0xffff) << 16;                // tile_dim0
  g1[4] = rows & 0xffff;                        // tile_dim1 (tile_dim2 = 0)
  g1[5] = stride_elts;                          // tensor_dim0_stride[31:0]
  g1[6] = 0;                                    // stride hi | dim1_stride lo
  g1[7] = 0;
  i32x4 z4 = {0, 0, 0, 0};
#if defined(__clang_major__) && __clang_major__ >= 23
  i32x8 z8 = {0, 0, 0, 0, 0, 0, 0, 0};
  __builtin_amdgcn_tensor_load_to_lds(g0, g1, z4, z4, z8, 0);
#else
  __builtin_amdgcn_tensor_load_to_lds(g0, g1, z4, z4, 0);
#endif
}

// ---------------------------------------------------------------------------
// f32 -> bf16 conversion (8 elements / thread, 32B loads)
// ---------------------------------------------------------------------------
__global__ void cvt_f32_bf16(const float* __restrict__ s,
                             bf16_t* __restrict__ d, int n) {
  int i = (blockIdx.x * blockDim.x + threadIdx.x) * 8;
  if (i + 8 <= n) {
    v8f x = *(const v8f*)(s + i);
    v8bf y;
#pragma unroll
    for (int j = 0; j < 8; ++j) y[j] = (bf16_t)x[j];
    *(v8bf*)(d + i) = y;
  }
}

// ---------------------------------------------------------------------------
// Projection GEMM:  Y[r][e] = sum_d A[r][d] * W[e][d] + bias[e]
// MODE 0: bf16 row-major out | MODE 1: bf16 transposed (Vt[b][e][n]) | MODE 2: f32 out
// Block: 256 thr = 8 waves; wave w -> cols [w*64, w*64+64); block -> 16 rows.
// ---------------------------------------------------------------------------
template <int MODE>
__global__ __launch_bounds__(256) void gemm_proj(
    const bf16_t* __restrict__ A, const bf16_t* __restrict__ W,
    const float* __restrict__ bias, bf16_t* __restrict__ Yb,
    float* __restrict__ Yf) {
  const int lane = threadIdx.x & 31;
  const int wave = threadIdx.x >> 5;
  const int hl = lane >> 4;       // lane-half select
  const int ln = lane & 15;
  const int rowbase = blockIdx.x * 16;

  v8f acc[4];
#pragma unroll
  for (int t = 0; t < 4; ++t) acc[t] = vzero8();

  const bf16_t* arow = A + (size_t)(rowbase + ln) * DIM;
#pragma unroll 4
  for (int kc = 0; kc < DIM / 32; ++kc) {
    // A fragment 16x32: lane row = ln; K = hl*16+0..15
    v8bf a0 = *(const v8bf*)(arow + kc * 32 + hl * 16);
    v8bf a1 = *(const v8bf*)(arow + kc * 32 + hl * 16 + 8);
    v16bf af = wmma_cat(a0, a1);
#pragma unroll
    for (int t = 0; t < 4; ++t) {
      int col = wave * 64 + t * 16 + ln;  // B col = lane n
      v16bf bfr = *(const v16bf*)(W + (size_t)col * DIM + kc * 32 + hl * 16);
      acc[t] = wmma_bf16(af, bfr, acc[t]);
    }
  }

#pragma unroll
  for (int t = 0; t < 4; ++t) {
    int col = wave * 64 + t * 16 + ln;
    float bv = bias[col];
#pragma unroll
    for (int i = 0; i < 8; ++i) {
      int row = rowbase + i + 8 * hl;   // C/D layout: M = i + 8*(lane/16)
      float y = acc[t][i] + bv;
      if (MODE == 2) {
        Yf[(size_t)row * DIM + col] = y;
      } else if (MODE == 0) {
        Yb[(size_t)row * DIM + col] = (bf16_t)y;
      } else {
        int b = row / SEQ, n = row % SEQ;
        Yb[((size_t)b * DIM + col) * SEQ + n] = (bf16_t)y;  // V transposed
      }
    }
  }
}

// ---------------------------------------------------------------------------
// Fused attention core:
//   O = softmax(Q K^T / sqrt(D)) @ V  +  intensity @ V     (bf16 out)
// Block: 16 queries, 8 waves, 32 keys/step.
//   - TDM (wave 0) streams the 16x32 intensity f32 tile into LDS, overlapped
//     with the score WMMAs; waited with s_wait_tensorcnt.
//   - waves split the d-reduction of the 16x32 score tile (2 chunks each);
//     deterministic LDS slab-reduce (no FP atomics).
//   - online softmax: each lane owns its fragment's 16 scores; row max/sum
//     completed with one shfl_xor(16) pair -> 16 exps/lane/step.
// ---------------------------------------------------------------------------
__global__ __launch_bounds__(256) void attn_core(
    const bf16_t* __restrict__ Q,   // [B*N][512]
    const bf16_t* __restrict__ K,   // [B*N][512]
    const bf16_t* __restrict__ Vt,  // [B][512][N]
    const float* __restrict__ inten,// [B][N][N]
    bf16_t* __restrict__ O) {       // [B*N][512]
  __shared__ float  s_red[8 * 16 * 32];            // 8 wave slabs (16 KB)
  __shared__ __align__(16) float  s_int[16 * 32];  // TDM intensity tile (2 KB)
  __shared__ __align__(16) bf16_t s_intb[16 * 32]; // bf16 intensity tile (1 KB)

  const int lane = threadIdx.x & 31;
  const int wave = threadIdx.x >> 5;
  const int hl = lane >> 4;
  const int ln = lane & 15;
  const int b = blockIdx.x >> 7;            // SEQ/16 = 128 query tiles
  const int qbase = (blockIdx.x & 127) * 16;
  const float scale = 0.044194173824159216f;  // 1/sqrt(512)

  v8f accS[4], accI[4];
#pragma unroll
  for (int t = 0; t < 4; ++t) { accS[t] = vzero8(); accI[t] = vzero8(); }

  float m_run = -__builtin_inff();  // running max for row = ln
  float l_run = 0.0f;               // running denom for row = ln

  const bf16_t* qrow = Q + ((size_t)b * SEQ + qbase + ln) * DIM;
  const float* itile = inten + ((size_t)b * SEQ + qbase) * SEQ;  // [16][SEQ]
  float* slab = s_red + wave * 512;
  const unsigned int_lds = (unsigned)(size_t)&s_int[0];

  for (int kb = 0; kb < SEQ; kb += 32) {
    // ---- 0) TDM: DMA this step's 16x32 intensity f32 tile into LDS --------
    if (wave == 0) tdm_load_2d_f32(itile + kb, int_lds, 16, 32, SEQ);

    // ---- 1) partial scores: wave sums d-chunks {wave, wave+8} -------------
    v8f sc[2];
    sc[0] = vzero8(); sc[1] = vzero8();
#pragma unroll
    for (int cp = 0; cp < 2; ++cp) {
      const int cc = wave + cp * 8;
      v8bf a0 = *(const v8bf*)(qrow + cc * 32 + hl * 16);
      v8bf a1 = *(const v8bf*)(qrow + cc * 32 + hl * 16 + 8);
      v16bf qf = wmma_cat(a0, a1);
#pragma unroll
      for (int t = 0; t < 2; ++t) {
        const bf16_t* krow =
            K + ((size_t)b * SEQ + kb + t * 16 + ln) * DIM + cc * 32 + hl * 16;
        v16bf kf = *(const v16bf*)krow;
        sc[t] = wmma_bf16(qf, kf, sc[t]);
      }
    }
    if (kb + 32 < SEQ)  // warm L2 for the next K tile
      __builtin_prefetch(K + ((size_t)b * SEQ + kb + 32 + ln) * DIM, 0, 1);

    // write this wave's 16x32 partial into its own slab (deterministic)
#pragma unroll
    for (int t = 0; t < 2; ++t)
#pragma unroll
      for (int i = 0; i < 8; ++i)
        slab[(i + 8 * hl) * 32 + t * 16 + ln] = sc[t][i];
    if (wave == 0) __builtin_amdgcn_s_wait_tensorcnt(0);  // intensity landed
    __syncthreads();

    // ---- 2) reduce 8 slabs -> slab 0; cvt intensity tile to bf16 ----------
    {
      const int e0 = threadIdx.x, e1 = threadIdx.x + 256;
      float s0 = 0.f, s1 = 0.f;
#pragma unroll
      for (int s = 0; s < 8; ++s) { s0 += s_red[s * 512 + e0]; s1 += s_red[s * 512 + e1]; }
      s_red[e0] = s0; s_red[e1] = s1;
      const int c0 = threadIdx.x * 2;
      s_intb[c0] = (bf16_t)s_int[c0];
      s_intb[c0 + 1] = (bf16_t)s_int[c0 + 1];
    }
    __syncthreads();

    // ---- 3) online softmax: lane owns its fragment's 16 scores ------------
    float sraw[16];
#pragma unroll
    for (int j = 0; j < 8; ++j) {
      sraw[j]     = s_red[ln * 32 + hl * 8 + j] * scale;
      sraw[j + 8] = s_red[ln * 32 + 16 + hl * 8 + j] * scale;
    }
    float mt = -__builtin_inff();
#pragma unroll
    for (int j = 0; j < 16; ++j) mt = fmaxf(mt, sraw[j]);
    mt = fmaxf(mt, __shfl_xor(mt, 16));            // combine lane halves
    const float m_new = fmaxf(m_run, mt);
    const float alpha = __expf(m_run - m_new);
    float p[16], psum = 0.0f;
#pragma unroll
    for (int j = 0; j < 16; ++j) { p[j] = __expf(sraw[j] - m_new); psum += p[j]; }
    psum += __shfl_xor(psum, 16);                  // full row sum
    l_run = l_run * alpha + psum;
    m_run = m_new;

    // rescale softmax accumulators (alpha per C-row via lane shuffle)
#pragma unroll
    for (int i = 0; i < 8; ++i) {
      const float ai = __shfl(alpha, i + 8 * hl);
#pragma unroll
      for (int t = 0; t < 4; ++t) accS[t][i] *= ai;
    }

    // ---- 4) P fragment: reuse the exp values directly ---------------------
    v8bf plo, phi;
#pragma unroll
    for (int j = 0; j < 8; ++j) { plo[j] = (bf16_t)p[j]; phi[j] = (bf16_t)p[j + 8]; }
    v16bf pf = wmma_cat(plo, phi);

    // ---- 5) intensity fragment straight from LDS (bf16) -------------------
    v8bf ilo = *(const v8bf*)(&s_intb[ln * 32 + hl * 8]);
    v8bf ihi = *(const v8bf*)(&s_intb[ln * 32 + 16 + hl * 8]);
    v16bf itf = wmma_cat(ilo, ihi);

    // ---- 6) shared V B-fragments feed both accumulators -------------------
#pragma unroll
    for (int t = 0; t < 4; ++t) {
      const int col = wave * 64 + t * 16 + ln;
      const bf16_t* vrow = Vt + ((size_t)b * DIM + col) * SEQ + kb + hl * 8;
      v16bf vf = wmma_cat(*(const v8bf*)vrow, *(const v8bf*)(vrow + 16));
      accS[t] = wmma_bf16(pf, vf, accS[t]);
      accI[t] = wmma_bf16(itf, vf, accI[t]);
    }
    __syncthreads();  // slabs + intensity consumed; safe to overwrite
  }

  // ---- epilogue: O = accS / l + accI ------------------------------------
  const float linv = 1.0f / l_run;  // for row = ln
#pragma unroll
  for (int t = 0; t < 4; ++t) {
    const int col = wave * 64 + t * 16 + ln;
#pragma unroll
    for (int i = 0; i < 8; ++i) {
      const int r = i + 8 * hl;
      const float li = __shfl(linv, r);
      const float y = accS[t][i] * li + accI[t][i];
      O[((size_t)b * SEQ + qbase + r) * DIM + col] = (bf16_t)y;
    }
  }
}

// ---------------------------------------------------------------------------
// Host launcher
// ---------------------------------------------------------------------------
extern "C" void kernel_launch(void* const* d_in, const int* in_sizes, int n_in,
                              void* d_out, int out_size, void* d_ws,
                              size_t ws_size, hipStream_t stream) {
  (void)in_sizes; (void)n_in; (void)out_size; (void)ws_size;
  const float* X     = (const float*)d_in[0];
  const float* inten = (const float*)d_in[1];
  const float* WQw = (const float*)d_in[2]; const float* WQb = (const float*)d_in[3];
  const float* WKw = (const float*)d_in[4]; const float* WKb = (const float*)d_in[5];
  const float* WVw = (const float*)d_in[6]; const float* WVb = (const float*)d_in[7];
  const float* WOw = (const float*)d_in[8]; const float* WOb = (const float*)d_in[9];

  char* ws = (char*)d_ws;
  size_t off = 0;
  auto carve = [&](size_t bytes) {
    void* p = ws + off;
    off = (off + bytes + 255) & ~(size_t)255;
    return p;
  };
  const size_t actN = (size_t)ROWS * DIM;      // 8.39M elements
  const size_t wN   = (size_t)DIM * DIM;       // 262144
  bf16_t* Xb  = (bf16_t*)carve(actN * 2);
  bf16_t* Qb  = (bf16_t*)carve(actN * 2);
  bf16_t* Kb  = (bf16_t*)carve(actN * 2);
  bf16_t* Vtb = (bf16_t*)carve(actN * 2);
  bf16_t* Ob  = (bf16_t*)carve(actN * 2);
  bf16_t* Wqb = (bf16_t*)carve(wN * 2);
  bf16_t* Wkb = (bf16_t*)carve(wN * 2);
  bf16_t* Wvb = (bf16_t*)carve(wN * 2);
  bf16_t* Wob = (bf16_t*)carve(wN * 2);

  // --- precision downconvert (bf16 feeds the WMMA pipes) ---
  cvt_f32_bf16<<<(int)(actN / 8 / 256), 256, 0, stream>>>(X, Xb, (int)actN);
  cvt_f32_bf16<<<(int)(wN / 8 / 256),  256, 0, stream>>>(WQw, Wqb, (int)wN);
  cvt_f32_bf16<<<(int)(wN / 8 / 256),  256, 0, stream>>>(WKw, Wkb, (int)wN);
  cvt_f32_bf16<<<(int)(wN / 8 / 256),  256, 0, stream>>>(WVw, Wvb, (int)wN);
  cvt_f32_bf16<<<(int)(wN / 8 / 256),  256, 0, stream>>>(WOw, Wob, (int)wN);

  const int gblocks = ROWS / 16;  // 1024
  gemm_proj<0><<<gblocks, 256, 0, stream>>>(Xb, Wqb, WQb, Qb, nullptr);
  gemm_proj<0><<<gblocks, 256, 0, stream>>>(Xb, Wkb, WKb, Kb, nullptr);
  gemm_proj<1><<<gblocks, 256, 0, stream>>>(Xb, Wvb, WVb, Vtb, nullptr);  // V^T

  attn_core<<<BATCH * (SEQ / 16), 256, 0, stream>>>(Qb, Kb, Vtb, inten, Ob);

  gemm_proj<2><<<gblocks, 256, 0, stream>>>(Ob, Wob, WOb, nullptr, (float*)d_out);
}
